// MY_BiLSTMCRF_Model_26018911879882
// MI455X (gfx1250) — matrix operand order repaired and checked by reading, not compiled, not run on hardware
//
#include <hip/hip_runtime.h>
#include <math.h>

#define Bn 64
#define Tn 512
#define En 128
#define Hn 256
#define Ln 32
#define G4H 1024
#define BT (Bn * Tn)
#define START_IDC 30
#define END_IDC 31

typedef __attribute__((ext_vector_type(16))) __bf16 v16bf;
typedef __attribute__((ext_vector_type(8)))  float  v8f;

union BF16Frag { v16bf v; uint4 q[2]; };

__device__ __forceinline__ unsigned short f2bf(float f) {
  unsigned int u = __float_as_uint(f);
  u += 0x7FFFu + ((u >> 16) & 1u);   // round-to-nearest-even
  return (unsigned short)(u >> 16);
}
__device__ __forceinline__ float bf2f(unsigned short s) {
  return __uint_as_float(((unsigned int)s) << 16);
}
__device__ __forceinline__ float sigf(float x) { return 1.0f / (1.0f + __expf(-x)); }

// ---------------------------------------------------------------------------
// fp32 -> bf16 conversion (weights)
// ---------------------------------------------------------------------------
__global__ void cvt_bf16_kernel(const float* __restrict__ src,
                                unsigned short* __restrict__ dst, int n) {
  int i = blockIdx.x * blockDim.x + threadIdx.x;
  if (i < n) dst[i] = f2bf(src[i]);
}

// ---------------------------------------------------------------------------
// embedding gather -> bf16, TIME-MAJOR rows: x[(t*B + b)*E + e]
// so each LSTM step's working set is one contiguous slice.
// ---------------------------------------------------------------------------
__global__ void embed_kernel(const float* __restrict__ emb,
                             const int* __restrict__ sent,
                             unsigned short* __restrict__ xbf) {
  int i = blockIdx.x * blockDim.x + threadIdx.x;
  if (i < BT * En) {
    int row = i >> 7;            // (t*B + b)
    int e   = i & (En - 1);
    int b   = row & (Bn - 1);
    int t   = row >> 6;
    xbf[i] = f2bf(emb[(size_t)sent[b * Tn + t] * En + e]);
  }
}

// ---------------------------------------------------------------------------
// mask transpose: maskT[t*B + b] = mask[b*T + t] (contiguous per-step reads)
// ---------------------------------------------------------------------------
__global__ void mask_transpose_kernel(const int* __restrict__ mask,
                                      int* __restrict__ maskT) {
  int i = blockIdx.x * blockDim.x + threadIdx.x;
  if (i < BT) {
    int b = i & (Bn - 1);
    int t = i >> 6;
    maskT[i] = mask[b * Tn + t];
  }
}

// ---------------------------------------------------------------------------
// Z = X(32768x128) @ Wih^T(128x1024) + bias  -- time-parallel input projection
// Flat rows are time-major (t*B+b); GEMM itself is layout-agnostic.
// grid: (BT/128, 1024/128, 2-directions), block 256 (8 waves)
// ---------------------------------------------------------------------------
__global__ void inproj_kernel(const unsigned short* __restrict__ xbf,
                              const unsigned short* __restrict__ WihF,
                              const unsigned short* __restrict__ WihB,
                              const float* __restrict__ bF,
                              const float* __restrict__ bB,
                              float* __restrict__ Zf, float* __restrict__ Zb) {
  const unsigned short* W = (blockIdx.z == 0) ? WihF : WihB;
  const float* bias       = (blockIdx.z == 0) ? bF : bB;
  float* Z                = (blockIdx.z == 0) ? Zf : Zb;

  const int lane = threadIdx.x & 31;
  const int wave = threadIdx.x >> 5;
  const int r0   = lane & 15;
  const int hi   = lane >> 4;
  const int selA = hi * 16;   // A-matrix K offset (lanes 16-31 hold K=16..31)
  const int selB = hi * 16;   // B-matrix K offset (lanes 16-31 hold K=16..31)
  const int rOff = hi * 8;    // D-matrix row offset

  const int mRow  = blockIdx.x * 128 + wave * 16;
  const int nBase = blockIdx.y * 128;
  const int aRow  = mRow + r0;

  // hoist A fragments (K = 128 -> 4 fragments of K=32)
  BF16Frag Af[4];
#pragma unroll
  for (int kk = 0; kk < 4; ++kk) {
    int k0 = kk * 32;
    Af[kk].q[0] = *(const uint4*)&xbf[(size_t)aRow * En + k0 + selA];
    Af[kk].q[1] = *(const uint4*)&xbf[(size_t)aRow * En + k0 + selA + 8];
  }

  for (int nt = 0; nt < 8; ++nt) {
    const int n = nBase + nt * 16 + r0;        // this lane's output column / W row
    float bv = bias[n];
    v8f acc;
#pragma unroll
    for (int r = 0; r < 8; ++r) acc[r] = bv;
#pragma unroll
    for (int kk = 0; kk < 4; ++kk) {
      int k0 = kk * 32;
      BF16Frag Bf;
      Bf.q[0] = *(const uint4*)&W[(size_t)n * En + k0 + selB];
      Bf.q[1] = *(const uint4*)&W[(size_t)n * En + k0 + selB + 8];
      acc = __builtin_amdgcn_wmma_f32_16x16x32_bf16(false, Af[kk].v, false, Bf.v,
                                                    (short)0, acc, false, false);
    }
#pragma unroll
    for (int r = 0; r < 8; ++r) {
      int row = mRow + rOff + r;
      Z[(size_t)row * G4H + n] = acc[r];
    }
  }
}

// ---------------------------------------------------------------------------
// Persistent bidirectional LSTM recurrence.
// grid 2 (dir), block 1024 (32 waves).  h carry: double-buffered LDS bf16.
// c carry: accumulator VGPRs.  Per step: 64x1024 = h(64x256)@Whh^T + Z[t].
// Z is time-major: step slice = contiguous 256KB; next slice prefetched
// (global_prefetch_b8) while current step's WMMA chain runs.
// ---------------------------------------------------------------------------
__global__ void lstm_kernel(const unsigned short* __restrict__ WhhF,
                            const unsigned short* __restrict__ WhhB,
                            const float* __restrict__ Zf,
                            const float* __restrict__ Zb,
                            const int* __restrict__ maskT,
                            unsigned short* __restrict__ hfOut,
                            unsigned short* __restrict__ hbOut) {
  const int dir = blockIdx.x;   // 0 = forward, 1 = backward
  const unsigned short* W = dir ? WhhB : WhhF;
  const float* Z          = dir ? Zb : Zf;
  unsigned short* hout    = dir ? hbOut : hfOut;

  __shared__ unsigned short hbuf[2][Bn * Hn];   // 2 x 32KB bf16 carry

  const int tid  = threadIdx.x;
  const int lane = tid & 31;
  const int wave = tid >> 5;
  const int r0   = lane & 15;
  const int hi   = lane >> 4;
  const int selA = hi * 16;
  const int selB = hi * 16;
  const int rOff = hi * 8;

  for (int i = tid; i < Bn * Hn; i += blockDim.x) hbuf[0][i] = 0;
  __syncthreads();

  v8f creg[2];
#pragma unroll
  for (int q = 0; q < 2; ++q)
#pragma unroll
    for (int r = 0; r < 8; ++r) creg[q][r] = 0.0f;

  for (int step = 0; step < Tn; ++step) {
    const int t   = dir ? (Tn - 1 - step) : step;
    const int cur = step & 1;
    const int nxt = cur ^ 1;

    // prefetch next step's Z slice (256KB = 2048 x 128B lines; 1024 thr x 2)
    const int tn = dir ? (t - 1) : (t + 1);
    if (tn >= 0 && tn < Tn) {
      const float* zn = Z + (size_t)tn * Bn * G4H;
      __builtin_prefetch(zn + (size_t)tid * 32, 0, 1);
      __builtin_prefetch(zn + (size_t)(tid + 1024) * 32, 0, 1);
    }

#pragma unroll
    for (int q = 0; q < 2; ++q) {
      const int p     = wave * 2 + q;
      const int mTile = p >> 4;          // 0..3  (16 batch rows each)
      const int jTile = p & 15;          // 0..15 (16 hidden cols each)
      const int colJ  = jTile * 16 + r0; // hidden index for this lane's column
      const int aRow  = mTile * 16 + r0;

      // accumulators seeded with the precomputed input projection Z[t,b,:]
      v8f acc[4];
#pragma unroll
      for (int g = 0; g < 4; ++g) {
        const int col = g * Hn + colJ;
#pragma unroll
        for (int r = 0; r < 8; ++r) {
          const int b = mTile * 16 + rOff + r;
          acc[g][r] = Z[((size_t)t * Bn + b) * G4H + col];
        }
      }

      for (int kk = 0; kk < 8; ++kk) {
        const int k0 = kk * 32;
        BF16Frag A;
        A.q[0] = *(const uint4*)&hbuf[cur][aRow * Hn + k0 + selA];
        A.q[1] = *(const uint4*)&hbuf[cur][aRow * Hn + k0 + selA + 8];
#pragma unroll
        for (int g = 0; g < 4; ++g) {
          const int n = g * Hn + colJ;   // Whh row = gate-major output column
          BF16Frag Bf;
          Bf.q[0] = *(const uint4*)&W[(size_t)n * Hn + k0 + selB];
          Bf.q[1] = *(const uint4*)&W[(size_t)n * Hn + k0 + selB + 8];
          acc[g] = __builtin_amdgcn_wmma_f32_16x16x32_bf16(false, A.v, false, Bf.v,
                                                           (short)0, acc[g], false, false);
        }
      }

      // fused gate nonlinearity + state update (i,f,g,o tiles all wave-local)
#pragma unroll
      for (int r = 0; r < 8; ++r) {
        const int b  = mTile * 16 + rOff + r;
        const float iv = sigf(acc[0][r]);
        const float fv = sigf(acc[1][r]);
        const float gv = tanhf(acc[2][r]);
        const float ov = sigf(acc[3][r]);
        const float cn = fv * creg[q][r] + iv * gv;
        const float hn = ov * tanhf(cn);
        const int   m  = maskT[t * Bn + b];
        float hcarry, ccarry;
        if (dir) {                       // reverse: zero outside sequence
          hcarry = m ? hn : 0.0f;
          ccarry = m ? cn : 0.0f;
        } else {                         // forward: hold previous state
          const float hold = bf2f(hbuf[cur][b * Hn + colJ]);
          hcarry = m ? hn : hold;
          ccarry = m ? cn : creg[q][r];
        }
        creg[q][r] = ccarry;
        hbuf[nxt][b * Hn + colJ] = f2bf(hcarry);
        hout[((size_t)t * Bn + b) * Hn + colJ] = f2bf(m ? hn : 0.0f);
      }
    }
    __syncthreads();   // writes to hbuf[nxt] visible before next step's reads
  }
}

// ---------------------------------------------------------------------------
// em = concat(hf, hb)(32768x512) @ Wc^T(512x32) + bc   (flat time-major rows)
// grid 256, block 256 (8 waves x 16 rows), 2 N-tiles per wave.
// ---------------------------------------------------------------------------
__global__ void classifier_kernel(const unsigned short* __restrict__ hf,
                                  const unsigned short* __restrict__ hb,
                                  const unsigned short* __restrict__ WcBf,
                                  const float* __restrict__ bc,
                                  float* __restrict__ em) {
  const int lane = threadIdx.x & 31;
  const int wave = threadIdx.x >> 5;
  const int r0   = lane & 15;
  const int hi   = lane >> 4;
  const int selA = hi * 16;
  const int selB = hi * 16;
  const int rOff = hi * 8;
  const int mRow = blockIdx.x * 128 + wave * 16;
  const int aRow = mRow + r0;

  v8f acc[2];
#pragma unroll
  for (int nt = 0; nt < 2; ++nt) {
    float bv = bc[nt * 16 + r0];
#pragma unroll
    for (int r = 0; r < 8; ++r) acc[nt][r] = bv;
  }

  for (int kk = 0; kk < 16; ++kk) {
    const int k0 = kk * 32;
    const unsigned short* src = (k0 < Hn) ? hf : hb;   // concat boundary is frag-aligned
    const int ko = k0 & (Hn - 1);
    BF16Frag A;
    A.q[0] = *(const uint4*)&src[(size_t)aRow * Hn + ko + selA];
    A.q[1] = *(const uint4*)&src[(size_t)aRow * Hn + ko + selA + 8];
#pragma unroll
    for (int nt = 0; nt < 2; ++nt) {
      const int n = nt * 16 + r0;
      BF16Frag Bf;
      Bf.q[0] = *(const uint4*)&WcBf[(size_t)n * (2 * Hn) + k0 + selB];
      Bf.q[1] = *(const uint4*)&WcBf[(size_t)n * (2 * Hn) + k0 + selB + 8];
      acc[nt] = __builtin_amdgcn_wmma_f32_16x16x32_bf16(false, A.v, false, Bf.v,
                                                        (short)0, acc[nt], false, false);
    }
  }
#pragma unroll
  for (int nt = 0; nt < 2; ++nt)
#pragma unroll
    for (int r = 0; r < 8; ++r)
      em[(size_t)(mRow + rOff + r) * Ln + nt * 16 + r0] = acc[nt][r];
}

// ---------------------------------------------------------------------------
// CRF gold path score: sum over masked (b,t) of em[t,b,tgt] + trans[prev,tgt]
// grid 64 (batch), block 256
// ---------------------------------------------------------------------------
__global__ void crf_gold_kernel(const float* __restrict__ em,
                                const int* __restrict__ targets,
                                const int* __restrict__ mask,
                                const float* __restrict__ trans,
                                float* __restrict__ acc) {
  const int b = blockIdx.x;
  float sum = 0.0f;
  for (int t = threadIdx.x; t < Tn; t += blockDim.x) {
    if (mask[b * Tn + t]) {
      const int tg = targets[b * Tn + t];
      const int pv = (t == 0) ? START_IDC : targets[b * Tn + t - 1];
      sum += em[((size_t)t * Bn + b) * Ln + tg] + trans[pv * Ln + tg];
    }
  }
  __shared__ float red[256];
  red[threadIdx.x] = sum;
  __syncthreads();
  for (int s = 128; s > 0; s >>= 1) {
    if (threadIdx.x < s) red[threadIdx.x] += red[threadIdx.x + s];
    __syncthreads();
  }
  if (threadIdx.x == 0) atomicAdd(&acc[1], red[0]);
}

// ---------------------------------------------------------------------------
// CRF forward algorithm: one wave32 per batch, lane j owns scores[j].
// grid 64, block 32
// ---------------------------------------------------------------------------
__global__ void crf_forward_kernel(const float* __restrict__ em,
                                   const int* __restrict__ mask,
                                   const float* __restrict__ trans,
                                   float* __restrict__ acc) {
  const int b = blockIdx.x;
  const int j = threadIdx.x;
  float tc[Ln];
#pragma unroll
  for (int i = 0; i < Ln; ++i) tc[i] = trans[i * Ln + j];

  float score = em[(size_t)b * Ln + j] + trans[START_IDC * Ln + j];   // t = 0 row

  for (int t = 1; t < Tn; ++t) {
    float mx = -INFINITY;
#pragma unroll
    for (int i = 0; i < Ln; ++i) {
      float v = __shfl(score, i, 32) + tc[i];
      mx = fmaxf(mx, v);
    }
    float s = 0.0f;
#pragma unroll
    for (int i = 0; i < Ln; ++i) {
      float v = __shfl(score, i, 32) + tc[i];
      s += __expf(v - mx);
    }
    const float nw = mx + __logf(s) + em[((size_t)t * Bn + b) * Ln + j];
    const int m = mask[b * Tn + t];
    score = m ? nw : score;
  }
  if (j == END_IDC) atomicAdd(&acc[0], score);
}

__global__ void init_acc_kernel(float* __restrict__ acc) {
  acc[0] = 0.0f;
  acc[1] = 0.0f;
}
__global__ void finalize_kernel(const float* __restrict__ acc, float* __restrict__ out) {
  out[0] = (acc[0] - acc[1]) / (float)Bn;
}

// ---------------------------------------------------------------------------
extern "C" void kernel_launch(void* const* d_in, const int* in_sizes, int n_in,
                              void* d_out, int out_size, void* d_ws, size_t ws_size,
                              hipStream_t stream) {
  (void)in_sizes; (void)n_in; (void)out_size; (void)ws_size;

  const int*   sentences = (const int*)d_in[0];
  const int*   mask      = (const int*)d_in[1];
  const int*   targets   = (const int*)d_in[2];
  const float* emb       = (const float*)d_in[3];
  const float* Wih_f     = (const float*)d_in[4];
  const float* Whh_f     = (const float*)d_in[5];
  const float* b_f       = (const float*)d_in[6];
  const float* Wih_b     = (const float*)d_in[7];
  const float* Whh_b     = (const float*)d_in[8];
  const float* b_b       = (const float*)d_in[9];
  const float* Wc        = (const float*)d_in[10];
  const float* bc        = (const float*)d_in[11];
  const float* trans     = (const float*)d_in[12];

  unsigned char* ws = (unsigned char*)d_ws;
  size_t o = 0;
  auto take = [&](size_t bytes) { size_t r = o; o += (bytes + 255) & ~(size_t)255; return r; };

  unsigned short* xbf    = (unsigned short*)(ws + take((size_t)BT * En * 2));
  unsigned short* WihFbf = (unsigned short*)(ws + take((size_t)G4H * En * 2));
  unsigned short* WihBbf = (unsigned short*)(ws + take((size_t)G4H * En * 2));
  unsigned short* WhhFbf = (unsigned short*)(ws + take((size_t)G4H * Hn * 2));
  unsigned short* WhhBbf = (unsigned short*)(ws + take((size_t)G4H * Hn * 2));
  unsigned short* WcBf   = (unsigned short*)(ws + take((size_t)Ln * 2 * Hn * 2));
  float*          Zf     = (float*)(ws + take((size_t)BT * G4H * 4));
  float*          Zb     = (float*)(ws + take((size_t)BT * G4H * 4));
  unsigned short* hfBuf  = (unsigned short*)(ws + take((size_t)BT * Hn * 2));
  unsigned short* hbBuf  = (unsigned short*)(ws + take((size_t)BT * Hn * 2));
  float*          em     = (float*)(ws + take((size_t)BT * Ln * 4));
  int*            maskT  = (int*)(ws + take((size_t)BT * 4));
  float*          acc    = (float*)(ws + take(2 * sizeof(float)));

  init_acc_kernel<<<1, 1, 0, stream>>>(acc);

  // weight conversion fp32 -> bf16
  cvt_bf16_kernel<<<(G4H * En + 255) / 256, 256, 0, stream>>>(Wih_f, WihFbf, G4H * En);
  cvt_bf16_kernel<<<(G4H * En + 255) / 256, 256, 0, stream>>>(Wih_b, WihBbf, G4H * En);
  cvt_bf16_kernel<<<(G4H * Hn + 255) / 256, 256, 0, stream>>>(Whh_f, WhhFbf, G4H * Hn);
  cvt_bf16_kernel<<<(G4H * Hn + 255) / 256, 256, 0, stream>>>(Whh_b, WhhBbf, G4H * Hn);
  cvt_bf16_kernel<<<(Ln * 2 * Hn + 255) / 256, 256, 0, stream>>>(Wc, WcBf, Ln * 2 * Hn);

  // embedding gather (time-major) + mask transpose
  embed_kernel<<<(BT * En + 255) / 256, 256, 0, stream>>>(emb, sentences, xbf);
  mask_transpose_kernel<<<(BT + 255) / 256, 256, 0, stream>>>(mask, maskT);

  // time-parallel input projections Z = X@Wih^T + b (both directions)
  inproj_kernel<<<dim3(BT / 128, G4H / 128, 2), 256, 0, stream>>>(
      xbf, WihFbf, WihBbf, b_f, b_b, Zf, Zb);

  // persistent bidirectional recurrence
  lstm_kernel<<<2, 1024, 0, stream>>>(WhhFbf, WhhBbf, Zf, Zb, maskT, hfBuf, hbBuf);

  // emission scores
  classifier_kernel<<<BT / 128, 256, 0, stream>>>(hfBuf, hbBuf, WcBf, bc, em);

  // CRF
  crf_gold_kernel<<<Bn, 256, 0, stream>>>(em, targets, mask, trans, acc);
  crf_forward_kernel<<<Bn, 32, 0, stream>>>(em, mask, trans, acc);
  finalize_kernel<<<1, 1, 0, stream>>>(acc, (float*)d_out);
}
